// CrossAttention_15917148799420
// MI455X (gfx1250) — matrix operand order repaired and checked by reading, not compiled, not run on hardware
//
#include <hip/hip_runtime.h>
#include <stdint.h>

typedef __bf16 bhalf;
typedef __attribute__((ext_vector_type(16))) __bf16 v16bf;
typedef __attribute__((ext_vector_type(8)))  float  v8f;

#define EMBED 1024
#define HEADS 16
#define HDIM  64
#define NB    2
#define SEQ   2048

#define BM 128
#define BN 128
#define BK 32
#define LDT 40   // padded LDS row stride for GEMM tiles (bf16 elems)

// softmax scale: 1/sqrt(1024) folded with log2(e) for exp2
#define SC2 (0.03125f * 1.4426950408889634f)

static __device__ __forceinline__ v8f wmma_bf16(v16bf a, v16bf b, v8f c) {
  return __builtin_amdgcn_wmma_f32_16x16x32_bf16(false, a, false, b, (short)0, c,
                                                 false, false);
}

static __device__ __forceinline__ void cvt16(const float4* s4, bhalf* dst) {
#pragma unroll
  for (int i = 0; i < 4; ++i) {
    float4 f = s4[i];
    dst[4*i+0] = (bhalf)f.x; dst[4*i+1] = (bhalf)f.y;
    dst[4*i+2] = (bhalf)f.z; dst[4*i+3] = (bhalf)f.w;
  }
}

// async copy: 16 bytes global -> LDS, per-lane addresses, tracked by ASYNCcnt
static __device__ __forceinline__ void async_b128(unsigned lds_off, const void* gptr) {
  asm volatile("global_load_async_to_lds_b128 %0, %1, off"
               :: "v"(lds_off), "v"(gptr) : "memory");
}

// ---------------------------------------------------------------------------
// C[M x 1024] = A(f32)[M x 1024] * W(f32)[1024 x 1024]^T
// HS=1: bf16 out, head-split     [N, H, L, 64]   (q, k)
// HS=2: bf16 out, head-transposed [N, H, 64, L]  (v -- async/WMMA friendly)
// HS=0: f32 out [M x 1024] + bias                (final projection)
// Double-buffered LDS, one barrier per K-step, global loads overlapped.
// ---------------------------------------------------------------------------
template<int HS>
__global__ __launch_bounds__(256) void gemm_xwT(
    const float* __restrict__ A, const float* __restrict__ W,
    bhalf* __restrict__ outHS, float* __restrict__ outF,
    const float* __restrict__ bias)
{
  __shared__ bhalf As[2][BM * LDT];
  __shared__ bhalf Bs[2][BN * LDT];

  const int bn = blockIdx.x, bm = blockIdx.y;
  const int tid = threadIdx.x;
  const int lane = tid & 31, wave = tid >> 5;
  const int wm = wave >> 1, wn = wave & 1;           // 4 x 2 wave grid
  const int lane15 = lane & 15, laneHalf = lane >> 4;

  const int frow = tid >> 1;            // 0..127
  const int fcb  = (tid & 1) * 16;      // 0 or 16

  const float4* aglob = (const float4*)(A + (size_t)(bm * BM + frow) * EMBED + fcb);
  const float4* bglob = (const float4*)(W + (size_t)(bn * BN + frow) * EMBED + fcb);

  v8f acc[2][4] = {};
  float4 ra[4], rb[4];

#pragma unroll
  for (int i = 0; i < 4; ++i) { ra[i] = aglob[i]; rb[i] = bglob[i]; }

  int buf = 0;
  for (int k0 = 0; k0 < EMBED; k0 += BK) {
    cvt16(ra, &As[buf][frow * LDT + fcb]);
    cvt16(rb, &Bs[buf][frow * LDT + fcb]);
    __syncthreads();

    if (k0 + BK < EMBED) {
      const float4* ag = aglob + (k0 + BK) / 4;
      const float4* bg = bglob + (k0 + BK) / 4;
#pragma unroll
      for (int i = 0; i < 4; ++i) { ra[i] = ag[i]; rb[i] = bg[i]; }
    }

    v16bf af[2], bfr[4];
#pragma unroll
    for (int mt = 0; mt < 2; ++mt) {
      const bhalf* base = &As[buf][(wm * 32 + mt * 16 + lane15) * LDT + laneHalf * 8];
      *((uint4*)&af[mt])     = *(const uint4*)(base);        // K 0..7 / 8..15
      *((uint4*)&af[mt] + 1) = *(const uint4*)(base + 16);   // K 16..23 / 24..31
    }
#pragma unroll
    for (int nt = 0; nt < 4; ++nt) {
      const bhalf* base = &Bs[buf][(wn * 64 + nt * 16 + lane15) * LDT + laneHalf * 8];
      *((uint4*)&bfr[nt])     = *(const uint4*)(base);
      *((uint4*)&bfr[nt] + 1) = *(const uint4*)(base + 16);
    }
#pragma unroll
    for (int mt = 0; mt < 2; ++mt)
#pragma unroll
      for (int nt = 0; nt < 4; ++nt)
        acc[mt][nt] = wmma_bf16(af[mt], bfr[nt], acc[mt][nt]);

    buf ^= 1;
  }

  // ---- writeback (C layout: elem i -> M = i + 8*laneHalf, N = lane15) ----
#pragma unroll
  for (int mt = 0; mt < 2; ++mt)
#pragma unroll
    for (int nt = 0; nt < 4; ++nt)
#pragma unroll
      for (int i = 0; i < 8; ++i) {
        int R = bm * BM + wm * 32 + mt * 16 + i + 8 * laneHalf;
        int C = bn * BN + wn * 64 + nt * 16 + lane15;
        float v = acc[mt][nt][i];
        if (HS == 1) {
          int n = R >> 11, l = R & (SEQ - 1);
          int h = C >> 6, d = C & 63;
          outHS[(((size_t)(n * HEADS + h) * SEQ) + l) * HDIM + d] = (bhalf)v;
        } else if (HS == 2) {
          int n = R >> 11, l = R & (SEQ - 1);
          int h = C >> 6, d = C & 63;
          outHS[(((size_t)(n * HEADS + h) * HDIM) + d) * SEQ + l] = (bhalf)v;
        } else {
          outF[(size_t)R * EMBED + C] = v + bias[C];
        }
      }
}

// ---------------------------------------------------------------------------
// C[n,h,d] = -1e20 * sum_{k : mask[n,k]==0} vprojT[n,h,d,k]
// ---------------------------------------------------------------------------
__global__ __launch_bounds__(64) void cconst_kernel(
    const bhalf* __restrict__ vprojT, const int* __restrict__ mask,
    float* __restrict__ cc)
{
  const int nh = blockIdx.x;            // 0..31
  const int d = threadIdx.x;            // 0..63
  const int n = nh >> 4, h = nh & 15;
  const bhalf* base = vprojT + ((size_t)nh * HDIM + d) * SEQ;   // contiguous row
  const int* mrow = mask + (size_t)n * SEQ;
  float s = 0.0f;
  for (int k = 0; k < SEQ; ++k)
    if (mrow[k] == 0) s += (float)base[k];
  cc[n * EMBED + h * HDIM + d] = -1e20f * s;
}

// ---------------------------------------------------------------------------
// Flash-style attention per (n, h, 128 q-rows). Mask-after-softmax quirk:
// masked columns stay in the denominator, are zeroed in the numerator, and the
// q-independent -1e20 constant C is added at the end.
// K [k][d] and V^T [d][k] tiles are DMA'd straight into LDS with
// GLOBAL_LOAD_ASYNC_TO_LDS_B128 (ASYNCcnt), double-buffered: tile i+1's
// transfers are issued right after the barrier and land during tile i compute.
// ---------------------------------------------------------------------------
#define KT 64
#define LDK 72   // padded LDS stride

__global__ __launch_bounds__(256) void attn_kernel(
    const bhalf* __restrict__ qp, const bhalf* __restrict__ kp,
    const bhalf* __restrict__ vpT, const int* __restrict__ mask,
    const float* __restrict__ cconst, float* __restrict__ attnout)
{
  __shared__ bhalf kt[2][KT * LDK];       // [key j][d]
  __shared__ bhalf vt[2][HDIM * LDK];     // [d][key j]
  __shared__ bhalf pt[8 * 16 * LDK];      // per-wave prob tiles [16][64]

  const int qt = blockIdx.x, h = blockIdx.y, n = blockIdx.z;
  const int tid = threadIdx.x;
  const int lane = tid & 31, wave = tid >> 5;
  const int lane15 = lane & 15, laneHalf = lane >> 4;
  const int qrow0 = qt * 128 + wave * 16;

  const size_t hbase = (size_t)(n * HEADS + h) * SEQ * HDIM;  // q/k and vT (same extent)
  const int* maskrow = mask + (size_t)n * SEQ;

  // per-thread fill mapping: row fr (0..63), 16 cols at fc0
  const int fr = tid >> 2, fc0 = (tid & 3) * 16;
  const bhalf* kgl = kp  + hbase + (size_t)fr * HDIM + fc0;   // K: row=key, col=d
  const bhalf* vgl = vpT + hbase + (size_t)fr * SEQ;          // V^T: row=d, col=key

  // q A-fragments for this wave's 16 rows (contraction d = 64 -> 2 frags)
  v16bf qa[2];
  {
    const bhalf* qb = qp + hbase + (size_t)(qrow0 + lane15) * HDIM;
#pragma unroll
    for (int f = 0; f < 2; ++f) {
      const bhalf* base = qb + f * 32 + laneHalf * 8;
      *((uint4*)&qa[f])     = *(const uint4*)(base);
      *((uint4*)&qa[f] + 1) = *(const uint4*)(base + 16);
    }
  }

  float m_r[8], l_r[8];
  v8f o[4] = {};
#pragma unroll
  for (int i = 0; i < 8; ++i) { m_r[i] = -3.0e38f; l_r[i] = 0.0f; }

  bhalf* pw = &pt[wave * 16 * LDK];

  // issue tile 0's async copies (4 x b128 per thread: 2 K + 2 V)
  {
    unsigned kl = (unsigned)(uintptr_t)&kt[0][fr * LDK + fc0];
    unsigned vl = (unsigned)(uintptr_t)&vt[0][fr * LDK + fc0];
    async_b128(kl,      kgl);
    async_b128(kl + 16, kgl + 8);
    async_b128(vl,      vgl + fc0);
    async_b128(vl + 16, vgl + fc0 + 8);
  }

  int buf = 0;
  for (int kb = 0; kb < SEQ; kb += KT) {
    // drain this wave's outstanding async copies (tile `buf`), then block-wide
    // barrier: everyone's copies are in LDS and everyone is done reading the
    // other buffer (barrier fence drains DScnt), so we may refill it.
    asm volatile("s_wait_asynccnt 0x0" ::: "memory");
    __syncthreads();

    if (kb + KT < SEQ) {
      const bhalf* ks = kgl + (size_t)(kb + KT) * HDIM;
      const bhalf* vs = vgl + (kb + KT) + fc0;
      unsigned kl = (unsigned)(uintptr_t)&kt[buf ^ 1][fr * LDK + fc0];
      unsigned vl = (unsigned)(uintptr_t)&vt[buf ^ 1][fr * LDK + fc0];
      async_b128(kl,      ks);
      async_b128(kl + 16, ks + 8);
      async_b128(vl,      vs);
      async_b128(vl + 16, vs + 8);
    }

    // mask values for this tile's columns (L2-hot, issued before the WMMAs)
    int mok4[4];
#pragma unroll
    for (int nt = 0; nt < 4; ++nt) mok4[nt] = maskrow[kb + nt * 16 + lane15];

    // ---- S = q . k^T  (16 x 64 per wave) ----
    v8f s[4] = {};
#pragma unroll
    for (int nt = 0; nt < 4; ++nt) {
      const int col = nt * 16 + lane15;
#pragma unroll
      for (int ks = 0; ks < 2; ++ks) {
        v16bf bfr;
        const bhalf* base = &kt[buf][col * LDK + ks * 32 + laneHalf * 8];
        *((uint4*)&bfr)     = *(const uint4*)(base);
        *((uint4*)&bfr + 1) = *(const uint4*)(base + 16);
        s[nt] = wmma_bf16(qa[ks], bfr, s[nt]);
      }
    }

    // ---- online softmax stats (rows live across lanes of one half) ----
    float corr[8];
#pragma unroll
    for (int i = 0; i < 8; ++i) {
      float mt_i = fmaxf(fmaxf(s[0][i], s[1][i]), fmaxf(s[2][i], s[3][i]));
#pragma unroll
      for (int off = 1; off < 16; off <<= 1)
        mt_i = fmaxf(mt_i, __shfl_xor(mt_i, off, 32));
      float mn = fmaxf(m_r[i], mt_i);
      corr[i] = exp2f((m_r[i] - mn) * SC2);
      m_r[i] = mn;
    }

    float lad[8] = {0, 0, 0, 0, 0, 0, 0, 0};
#pragma unroll
    for (int nt = 0; nt < 4; ++nt) {
#pragma unroll
      for (int i = 0; i < 8; ++i) {
        float p = exp2f((s[nt][i] - m_r[i]) * SC2);
        lad[i] += p;                              // denominator: ALL columns
        float pm = mok4[nt] ? p : 0.0f;           // numerator: masked zeroed
        pw[(i + 8 * laneHalf) * LDK + nt * 16 + lane15] = (bhalf)pm;
      }
    }
#pragma unroll
    for (int i = 0; i < 8; ++i) {
      float ls = lad[i];
#pragma unroll
      for (int off = 1; off < 16; off <<= 1) ls += __shfl_xor(ls, off, 32);
      l_r[i] = l_r[i] * corr[i] + ls;
#pragma unroll
      for (int t = 0; t < 4; ++t) o[t][i] *= corr[i];
    }

    // P-tile is produced and consumed by the SAME wave; per-wave LDS ops are
    // in-order (DScnt) and the compiler inserts s_wait_dscnt before reuse.
    asm volatile("" ::: "memory");

    // ---- O += P . V ----
#pragma unroll
    for (int ks = 0; ks < 2; ++ks) {
      v16bf pa;
      const bhalf* base = &pw[lane15 * LDK + ks * 32 + laneHalf * 8];
      *((uint4*)&pa)     = *(const uint4*)(base);
      *((uint4*)&pa + 1) = *(const uint4*)(base + 16);
#pragma unroll
      for (int t = 0; t < 4; ++t) {
        v16bf vb;
        const bhalf* vbase = &vt[buf][(t * 16 + lane15) * LDK + ks * 32 + laneHalf * 8];
        *((uint4*)&vb)     = *(const uint4*)(vbase);
        *((uint4*)&vb + 1) = *(const uint4*)(vbase + 16);
        o[t] = wmma_bf16(pa, vb, o[t]);
      }
    }

    buf ^= 1;   // next iteration consumes the buffer being filled now
  }

  // ---- epilogue: acc/l + mask constant, write f32 [N, Q, EMBED] ----
  const float* ccp = cconst + n * EMBED + h * HDIM;
#pragma unroll
  for (int t = 0; t < 4; ++t) {
    const int d = t * 16 + lane15;
    const float cc = ccp[d];
#pragma unroll
    for (int i = 0; i < 8; ++i) {
      int qr = qrow0 + i + 8 * laneHalf;
      attnout[((size_t)n * SEQ + qr) * EMBED + h * HDIM + d] = o[t][i] / l_r[i] + cc;
    }
  }
}

// ---------------------------------------------------------------------------
extern "C" void kernel_launch(void* const* d_in, const int* in_sizes, int n_in,
                              void* d_out, int out_size, void* d_ws, size_t ws_size,
                              hipStream_t stream) {
  (void)in_sizes; (void)n_in; (void)out_size; (void)ws_size;
  const float* dec   = (const float*)d_in[0];  // [N,Q,1024]
  const float* enc_k = (const float*)d_in[1];  // [N,K,1024]
  const float* enc_v = (const float*)d_in[2];  // [N,K,1024]
  const int*   mask  = (const int*)  d_in[3];  // [N,1,1,K]
  const float* Wv    = (const float*)d_in[4];
  const float* Wk    = (const float*)d_in[5];
  const float* Wq    = (const float*)d_in[6];
  const float* Wo    = (const float*)d_in[7];
  const float* bo    = (const float*)d_in[8];

  const size_t PROJ = (size_t)NB * HEADS * SEQ * HDIM;   // 4M bf16 elems
  bhalf* qproj  = (bhalf*)d_ws;
  bhalf* kproj  = qproj + PROJ;
  bhalf* vprojT = kproj + PROJ;                          // [N,H,64,SEQ]
  float* attnout = (float*)(vprojT + PROJ);
  float* cconst  = attnout + (size_t)NB * SEQ * EMBED;

  dim3 gproj(EMBED / BN, (NB * SEQ) / BM);   // (8, 32)
  gemm_xwT<1><<<gproj, 256, 0, stream>>>(dec,   Wq, qproj,  nullptr, nullptr);
  gemm_xwT<1><<<gproj, 256, 0, stream>>>(enc_k, Wk, kproj,  nullptr, nullptr);
  gemm_xwT<2><<<gproj, 256, 0, stream>>>(enc_v, Wv, vprojT, nullptr, nullptr);

  cconst_kernel<<<NB * HEADS, 64, 0, stream>>>(vprojT, mask, cconst);

  attn_kernel<<<dim3(SEQ / 128, HEADS, NB), 256, 0, stream>>>(
      qproj, kproj, vprojT, mask, cconst, attnout);

  gemm_xwT<0><<<gproj, 256, 0, stream>>>(attnout, Wo, nullptr, (float*)d_out, bo);
}
